// SimpleTwoLayerNet_68118181315033
// MI455X (gfx1250) — matrix operand order repaired and checked by reading, not compiled, not run on hardware
//
#include <hip/hip_runtime.h>
#include <hip/hip_bf16.h>

typedef __attribute__((ext_vector_type(16))) __bf16 bf16x16;
typedef __attribute__((ext_vector_type(8)))  float  floatx8;

union Frag32 {
    uint4   q[2];
    bf16x16 v;
};

__device__ __forceinline__ unsigned short f2bf(float f) {
    unsigned int u = __float_as_uint(f);
    unsigned int r = u + 0x7fffu + ((u >> 16) & 1u);   // round-to-nearest-even
    return (unsigned short)(r >> 16);
}

__device__ __forceinline__ float fast_rcp(float f) {
    return __builtin_amdgcn_rcpf(f);                   // single v_rcp_f32
}

__device__ __forceinline__ floatx8 wmma_bf16(const Frag32& a, const Frag32& b, floatx8 c) {
    return __builtin_amdgcn_wmma_f32_16x16x32_bf16(
        /*neg_a=*/false, a.v, /*neg_b=*/false, b.v,
        /*c_mod=*/(short)0, c, /*reuse_a=*/false, /*reuse_b=*/false);
}

#define B_ROWS   8192
#define D_DIM    512
#define P_DIM    4096
#define C_OUT    1000
#define C_PAD    1024
#define CHUNK    1024
#define NCHUNK   (P_DIM / CHUNK)

// LDS pitches (in 16-bit elements), padded by 8 bf16 (4 words) to skew banks
#define XPITCH_H    (D_DIM + 8)      // 520 bf16 -> 260 words, 260%64==4
#define APITCH_H    (CHUNK + 8)      // 1032 bf16 -> 516 words, 516%64==4
#define XPITCH_W    (XPITCH_H / 2)   // 260
#define APITCH_W    (APITCH_H / 2)   // 516

// ---------------------------------------------------------------------------
// Kernel 1: keys f32 -> bf16, plus per-key squared norms. One wave per key row.
// ---------------------------------------------------------------------------
__global__ __launch_bounds__(256) void convert_keys_kernel(
    const float* __restrict__ keys, unsigned short* __restrict__ kbf,
    float* __restrict__ kn)
{
    int wave = threadIdx.x >> 5;
    int lane = threadIdx.x & 31;
    int row  = blockIdx.x * 8 + wave;          // 512 blocks * 8 waves = 4096 rows
    const float* kr = keys + (size_t)row * D_DIM;
    unsigned short* ko = kbf + (size_t)row * D_DIM;
    float s = 0.f;
    #pragma unroll
    for (int i = lane; i < D_DIM; i += 32) {
        float v = kr[i];
        ko[i] = f2bf(v);
        s += v * v;
    }
    #pragma unroll
    for (int m = 16; m >= 1; m >>= 1) s += __shfl_xor(s, m, 32);
    if (lane == 0) kn[row] = s;
}

// ---------------------------------------------------------------------------
// Kernel 2: values [4096][1000] f32 -> values_t [1024][4096] bf16 (transposed,
// zero-padded columns 1000..1023). 32x32 tiles through LDS.
// ---------------------------------------------------------------------------
__global__ __launch_bounds__(256) void transpose_values_kernel(
    const float* __restrict__ vals, unsigned short* __restrict__ vt)
{
    __shared__ unsigned short tile[32][33];
    int tx = threadIdx.x & 31;
    int ty = threadIdx.x >> 5;                 // 0..7
    int pb = blockIdx.x * 32;                  // along P (4096/32 = 128)
    int cb = blockIdx.y * 32;                  // along C (1024/32 = 32)
    #pragma unroll
    for (int i = 0; i < 4; i++) {
        int p = pb + ty + i * 8;
        int c = cb + tx;
        float v = (c < C_OUT) ? vals[(size_t)p * C_OUT + c] : 0.f;
        tile[ty + i * 8][tx] = f2bf(v);
    }
    __syncthreads();
    #pragma unroll
    for (int i = 0; i < 4; i++) {
        int c = cb + ty + i * 8;
        int p = pb + tx;
        vt[(size_t)c * P_DIM + p] = tile[tx][ty + i * 8];
    }
}

// ---------------------------------------------------------------------------
// Kernel 3: fused distance -> inverse-power attention -> attn @ values.
// One workgroup = 16 x-rows, all 1024 (padded) output columns.
// ---------------------------------------------------------------------------
__global__ __launch_bounds__(256) void fused_attn_kernel(
    const float* __restrict__ x,
    const unsigned short* __restrict__ kbf,
    const float* __restrict__ kn,
    const unsigned short* __restrict__ vt,
    float* __restrict__ out)
{
    __shared__ unsigned int  xlds[16 * XPITCH_W];   // 16,640 B
    __shared__ unsigned int  awrd[16 * APITCH_W];   // 33,024 B
    __shared__ float         xns[16];

    unsigned short* ah = (unsigned short*)awrd;
    unsigned short* xh = (unsigned short*)xlds;

    const int tid  = threadIdx.x;
    const int wave = tid >> 5;
    const int lane = tid & 31;
    const int lrow = lane & 15;       // fragment row / col-in-tile index
    const int lhi  = lane >> 4;       // 0 or 1 (lane half)
    const int row0 = blockIdx.x * 16; // global x-row base

    if (tid < 16) xns[tid] = 0.f;
    __syncthreads();

    // ---- stage x tile: 16 rows x 512 f32 -> bf16 in LDS, accumulate ||x||^2
    {
        int srow = tid >> 4;               // 0..15
        int scol = (tid & 15) * 32;        // 32 contiguous elems per thread
        const float4* xr = (const float4*)(x + (size_t)(row0 + srow) * D_DIM + scol);
        float ssq = 0.f;
        #pragma unroll
        for (int i = 0; i < 8; i++) {
            float4 v = xr[i];
            ssq += v.x * v.x + v.y * v.y + v.z * v.z + v.w * v.w;
            int c = scol + i * 4;
            xh[srow * XPITCH_H + c + 0] = f2bf(v.x);
            xh[srow * XPITCH_H + c + 1] = f2bf(v.y);
            xh[srow * XPITCH_H + c + 2] = f2bf(v.z);
            xh[srow * XPITCH_H + c + 3] = f2bf(v.w);
        }
        atomicAdd(&xns[srow], ssq);
    }
    __syncthreads();

    float xnr[8];
    #pragma unroll
    for (int i = 0; i < 8; i++) xnr[i] = xns[i + 8 * lhi];

    float ps[8] = {0.f};                  // rowsum partials
    floatx8 accv[8];
    #pragma unroll
    for (int j = 0; j < 8; j++)
        #pragma unroll
        for (int i = 0; i < 8; i++) accv[j][i] = 0.f;

    for (int chunk = 0; chunk < NCHUNK; chunk++) {
        const int pbase = chunk * CHUNK;

        // ---- Stage A: this wave computes attn weights for p-range of 128.
        // Two interleaved accumulator chains (even/odd K-steps) for WMMA ILP.
        for (int t = 0; t < 8; t++) {
            const int p0 = pbase + wave * 128 + t * 16;
            floatx8 dacc0, dacc1;
            #pragma unroll
            for (int i = 0; i < 8; i++) { dacc0[i] = 0.f; dacc1[i] = 0.f; }

            #pragma unroll 2
            for (int ks = 0; ks < 8; ks++) {
                const int kb0 = (2 * ks) * 32;
                const int kb1 = (2 * ks + 1) * 32;
                Frag32 a0, b0, a1, b1;
                // A fragments from LDS x tile (16-bit A 16x32 layout)
                int w0 = lrow * XPITCH_W + (kb0 >> 1) + lhi * 8;
                int w1 = lrow * XPITCH_W + (kb1 >> 1) + lhi * 8;
                a0.q[0] = *(const uint4*)(&xlds[w0]);
                a0.q[1] = *(const uint4*)(&xlds[w0 + 4]);
                a1.q[0] = *(const uint4*)(&xlds[w1]);
                a1.q[1] = *(const uint4*)(&xlds[w1 + 4]);
                // B fragments from global keys_bf16 (B 32x16 layout)
                const uint4* bp0 = (const uint4*)(kbf + (size_t)(p0 + lrow) * D_DIM
                                                  + kb0 + lhi * 16);
                const uint4* bp1 = (const uint4*)(kbf + (size_t)(p0 + lrow) * D_DIM
                                                  + kb1 + lhi * 16);
                b0.q[0] = bp0[0];
                b0.q[1] = bp0[1];
                b1.q[0] = bp1[0];
                b1.q[1] = bp1[1];
                dacc0 = wmma_bf16(a0, b0, dacc0);
                dacc1 = wmma_bf16(a1, b1, dacc1);
            }

            const float knv = kn[p0 + lrow];
            const int   col = wave * 128 + t * 16 + lrow;   // col within chunk
            #pragma unroll
            for (int i = 0; i < 8; i++) {
                float dot = dacc0[i] + dacc1[i];
                float d2 = xnr[i] + knv - 2.0f * dot;
                d2 = fmaxf(d2, 0.f);
                float w = fast_rcp(1e-3f + d2);             // EPS + d^POWER, POWER=2
                ps[i] += w;
                ah[(i + 8 * lhi) * APITCH_H + col] = f2bf(w);
            }
        }
        __syncthreads();

        // ---- Stage B: attn(chunk) @ values_t; wave owns 128 output cols.
        // accv[0..7] are 8 independent WMMA chains sharing one A fragment.
        for (int kk = 0; kk < 32; kk++) {
            const int kb = kk * 32;
            Frag32 a;
            int w0 = lrow * APITCH_W + (kb >> 1) + lhi * 8;
            a.q[0] = *(const uint4*)(&awrd[w0]);
            a.q[1] = *(const uint4*)(&awrd[w0 + 4]);
            const int p = pbase + kb + lhi * 16;
            #pragma unroll
            for (int j = 0; j < 8; j++) {
                Frag32 b;
                const int c = wave * 128 + j * 16 + lrow;
                const uint4* bp = (const uint4*)(vt + (size_t)c * P_DIM + p);
                b.q[0] = bp[0];
                b.q[1] = bp[1];
                accv[j] = wmma_bf16(a, b, accv[j]);
            }
        }
        __syncthreads();
    }

    // ---- rowsum reduction and final scaled store
    if (tid < 16) xns[tid] = 0.f;
    __syncthreads();
    #pragma unroll
    for (int i = 0; i < 8; i++) atomicAdd(&xns[i + 8 * lhi], ps[i]);
    __syncthreads();

    float inv[8];
    #pragma unroll
    for (int i = 0; i < 8; i++) inv[i] = fast_rcp(xns[i + 8 * lhi]);

    #pragma unroll
    for (int j = 0; j < 8; j++) {
        const int c = wave * 128 + j * 16 + lrow;
        if (c < C_OUT) {
            #pragma unroll
            for (int i = 0; i < 8; i++) {
                const int r = row0 + i + 8 * lhi;
                out[(size_t)r * C_OUT + c] = accv[j][i] * inv[i];
            }
        }
    }
}

// ---------------------------------------------------------------------------
extern "C" void kernel_launch(void* const* d_in, const int* in_sizes, int n_in,
                              void* d_out, int out_size, void* d_ws, size_t ws_size,
                              hipStream_t stream) {
    (void)in_sizes; (void)n_in; (void)out_size; (void)ws_size;
    const float* x    = (const float*)d_in[0];   // [8192,512]
    const float* keys = (const float*)d_in[1];   // [4096,512]
    const float* vals = (const float*)d_in[2];   // [4096,1000]
    float* out = (float*)d_out;                  // [8192,1000]

    // Workspace layout
    unsigned short* kbf = (unsigned short*)d_ws;                               // 4 MB
    float*          kn  = (float*)((char*)d_ws + 4u * 1024u * 1024u);          // 16 KB
    unsigned short* vt  = (unsigned short*)((char*)d_ws + 4u * 1024u * 1024u
                                            + 64u * 1024u);                    // 8 MB

    convert_keys_kernel<<<P_DIM / 8, 256, 0, stream>>>(keys, kbf, kn);
    transpose_values_kernel<<<dim3(P_DIM / 32, C_PAD / 32), 256, 0, stream>>>(vals, vt);
    fused_attn_kernel<<<B_ROWS / 16, 256, 0, stream>>>(x, kbf, kn, vt, out);
}
